// SimpleAttention_61065845015130
// MI455X (gfx1250) — matrix operand order repaired and checked by reading, not compiled, not run on hardware
//
#include <hip/hip_runtime.h>
#include <hip/hip_bf16.h>

typedef __attribute__((ext_vector_type(16))) _Float16 v16h;
typedef __attribute__((ext_vector_type(4)))  _Float16 v4h;
typedef __attribute__((ext_vector_type(8)))  float    v8f;

constexpr int kDim   = 1024;
constexpr int kHeads = 16;
constexpr int kHd    = 64;
constexpr int kB     = 4;
constexpr int kL     = 2048;
constexpr int kM     = kB * kL;   // 8192 token rows

#define NEG_INF (-__builtin_inff())

// ---------------------------------------------------------------------------
// CDNA5 async global->LDS copy (ASYNCcnt-tracked), per CDNA5 ISA.
// Each lane moves 16 bytes: LDS[vdst_lane] = MEM[vaddr_lane].
// Generic LDS pointer low 32 bits == LDS byte address (aperture rule).
// ---------------------------------------------------------------------------
__device__ __forceinline__ void async_b128(const _Float16* g, _Float16* l) {
  unsigned lds_addr = (unsigned)(size_t)l;
  unsigned long long gaddr = (unsigned long long)(size_t)g;
  asm volatile("global_load_async_to_lds_b128 %0, %1, off"
               :: "v"(lds_addr), "v"(gaddr)
               : "memory");
}
#define S_WAIT_ASYNC(n) asm volatile("s_wait_asynccnt " #n ::: "memory")

// ---------------------------------------------------------------------------
// WMMA helpers (CDNA5 wave32, 16x16x32 f16 -> f32)
// ---------------------------------------------------------------------------
__device__ __forceinline__ v8f wmma16(v16h a, v16h b, v8f c) {
  return __builtin_amdgcn_wmma_f32_16x16x32_f16(false, a, false, b, (short)0, c,
                                                false, false);
}

union FragU {
  v16h v;
  uint4 u[2];
  _Float16 h[16];
};

// A-matrix 16x32 f16 fragment from row-major LDS tile (ldh halves per row).
// lanes 0-15: M=lane,  halves = K[0..7],  K[16..23]
// lanes 16-31: M=lane-16, halves = K[8..15], K[24..31]
__device__ __forceinline__ v16h frag_a_lds(const _Float16* base, int ldh, int lane) {
  FragU f;
  const _Float16* r = base + (lane & 15) * ldh + ((lane < 16) ? 0 : 8);
  f.u[0] = *(const uint4*)(r);
  f.u[1] = *(const uint4*)(r + 16);
  return f.v;
}

// B-matrix 32x16 f16 fragment from LDS tile stored N-major (ldh halves per N row).
// lanes 0-15: N=lane, halves = K[0..7], K[16..23]; lanes 16-31: N=lane-16, K[8..15], K[24..31]
__device__ __forceinline__ v16h frag_b_lds(const _Float16* base, int ldh, int lane) {
  FragU f;
  const _Float16* r = base + (lane & 15) * ldh + ((lane < 16) ? 0 : 8);
  f.u[0] = *(const uint4*)(r);
  f.u[1] = *(const uint4*)(r + 16);
  return f.v;
}

// ---------------------------------------------------------------------------
// fp32 -> f16 conversion (4-wide)
// ---------------------------------------------------------------------------
__global__ void cvt_f32_f16(const float* __restrict__ in,
                            _Float16* __restrict__ out, int n4) {
  int i = blockIdx.x * blockDim.x + threadIdx.x;
  if (i < n4) {
    float4 f = ((const float4*)in)[i];
    v4h o;
    o[0] = (_Float16)f.x; o[1] = (_Float16)f.y;
    o[2] = (_Float16)f.z; o[3] = (_Float16)f.w;
    ((v4h*)out)[i] = o;
  }
}

// ---------------------------------------------------------------------------
// GEMM: C[M,N] = A[M,K] * B[N,K]^T   (A,B f16 row-major; accum f32)
// Block tile 128x128, BK=32, 256 threads = 8 waves in 4x2; wave tile 32x64.
// Double-buffered LDS fed by global_load_async_to_lds_b128:
//   prologue: issue tile0 (4 async/thread)
//   iter kt : issue tile kt+1 into other buffer, s_wait_asynccnt 4 -> tile kt
//             resident, barrier, 8 WMMAs, barrier.
// MODE 0: store f32 C row-major.  MODE 1: scatter qkv -> [B,H,L,Hd] f16.
// ---------------------------------------------------------------------------
template <int MODE>
__global__ void __launch_bounds__(256)
gemm_wmma(const _Float16* __restrict__ A, const _Float16* __restrict__ B,
          float* __restrict__ C, _Float16* __restrict__ qo,
          _Float16* __restrict__ ko, _Float16* __restrict__ vo,
          int M, int N, int K) {
  __shared__ __attribute__((aligned(16))) _Float16 As[2][128 * 32];
  __shared__ __attribute__((aligned(16))) _Float16 Bs[2][128 * 32];

  const int tid  = threadIdx.x;
  const int lane = tid & 31;
  const int w    = tid >> 5;   // 0..7
  const int wm   = w & 3;      // 0..3 -> 32-row group
  const int wn   = w >> 2;     // 0..1 -> 64-col group
  const int bm   = blockIdx.y * 128;
  const int bn   = blockIdx.x * 128;

  v8f acc[2][4];
#pragma unroll
  for (int mi = 0; mi < 2; ++mi)
#pragma unroll
    for (int ni = 0; ni < 4; ++ni) acc[mi][ni] = {};

  // cooperative tile copy: 2 threads per row, 32B (2 x b128) each of A and B
  const int lrow = tid >> 1;            // 0..127
  const int lcol = (tid & 1) * 16;      // halves
  const _Float16* Ag = A + (size_t)(bm + lrow) * K + lcol;
  const _Float16* Bg = B + (size_t)(bn + lrow) * K + lcol;
  const int ldsoff = lrow * 32 + lcol;

  const int nk = K >> 5;  // K/32 tiles

  // prologue: tile 0 -> buffer 0
  async_b128(Ag, &As[0][ldsoff]);
  async_b128(Ag + 8, &As[0][ldsoff + 8]);
  async_b128(Bg, &Bs[0][ldsoff]);
  async_b128(Bg + 8, &Bs[0][ldsoff + 8]);

  for (int kt = 0; kt < nk; ++kt) {
    const int cur = kt & 1;
    if (kt + 1 < nk) {
      const int k0 = (kt + 1) << 5;
      const int nxt = cur ^ 1;
      async_b128(Ag + k0, &As[nxt][ldsoff]);
      async_b128(Ag + k0 + 8, &As[nxt][ldsoff + 8]);
      async_b128(Bg + k0, &Bs[nxt][ldsoff]);
      async_b128(Bg + k0 + 8, &Bs[nxt][ldsoff + 8]);
      S_WAIT_ASYNC(0x4);   // current tile's 4 loads complete (in-order)
    } else {
      S_WAIT_ASYNC(0x0);
    }
    __syncthreads();

    v16h af[2], bf[4];
    const _Float16* abase = &As[cur][(wm * 32) * 32];
    af[0] = frag_a_lds(abase, 32, lane);
    af[1] = frag_a_lds(abase + 16 * 32, 32, lane);
#pragma unroll
    for (int ni = 0; ni < 4; ++ni)
      bf[ni] = frag_b_lds(&Bs[cur][(wn * 64 + ni * 16) * 32], 32, lane);

#pragma unroll
    for (int mi = 0; mi < 2; ++mi)
#pragma unroll
      for (int ni = 0; ni < 4; ++ni)
        acc[mi][ni] = wmma16(af[mi], bf[ni], acc[mi][ni]);
    __syncthreads();  // readers done before next iteration reuses this buffer
  }

  // epilogue: C/D layout -> row = vgpr + 8*(lane/16), col = lane%16
  const int g = lane >> 4, c = lane & 15;
#pragma unroll
  for (int mi = 0; mi < 2; ++mi) {
#pragma unroll
    for (int ni = 0; ni < 4; ++ni) {
      const int row0 = bm + wm * 32 + mi * 16 + 8 * g;
      const int col  = bn + wn * 64 + ni * 16 + c;
#pragma unroll
      for (int j = 0; j < 8; ++j) {
        const int row = row0 + j;
        if (MODE == 0) {
          C[(size_t)row * N + col] = acc[mi][ni][j];
        } else {
          const int seg = col >> 10;       // 0=q 1=k 2=v
          const int nc  = col & 1023;
          const int hh  = nc >> 6;
          const int hd  = nc & 63;
          const int bb  = row >> 11;
          const int ll  = row & 2047;
          _Float16* dst = (seg == 0) ? qo : (seg == 1) ? ko : vo;
          dst[(((size_t)bb * kHeads + hh) * kL + ll) * kHd + hd] =
              (_Float16)acc[mi][ni][j];
        }
      }
    }
  }
}

// ---------------------------------------------------------------------------
// Flash attention (causal). One block (4 waves, 128 thr) per (b, h, 64-row Qtile).
// q/k/v: f16 [B,H,L,Hd]. Output: f16 [B,L,Dim].
// Q and K tiles staged via async-to-LDS; V transposed through registers.
// ---------------------------------------------------------------------------
__global__ void __launch_bounds__(128)
flash_attn(const _Float16* __restrict__ Q, const _Float16* __restrict__ Kk,
           const _Float16* __restrict__ V, _Float16* __restrict__ Oo) {
  __shared__ __attribute__((aligned(16))) _Float16 Qs[64 * 64];
  __shared__ __attribute__((aligned(16))) _Float16 Ks[64 * 64];
  __shared__ __attribute__((aligned(16))) _Float16 Vt[64 * 64];  // [hd][key]
  __shared__ __attribute__((aligned(16))) _Float16 Ps[64 * 64];

  const int tid  = threadIdx.x;
  const int lane = tid & 31;
  const int w    = tid >> 5;       // 0..3 -> 16-row group
  const int qb   = blockIdx.x;     // 0..31
  const int h    = blockIdx.y;     // 0..15
  const int b    = blockIdx.z;     // 0..3
  const int g    = lane >> 4, c = lane & 15;

  const size_t headoff = ((size_t)b * kHeads + h) * kL * kHd;

  // Q tile (64 rows x 128B): 2 threads/row, 64B = 4 async b128 each
  const int lqrow = tid >> 1;
  const int lqco  = (tid & 1) * 32;
  {
    const _Float16* src = Q + headoff + ((size_t)qb * 64 + lqrow) * kHd + lqco;
    _Float16* dst = Qs + lqrow * 64 + lqco;
    async_b128(src, dst);
    async_b128(src + 8, dst + 8);
    async_b128(src + 16, dst + 16);
    async_b128(src + 24, dst + 24);
  }

  v8f o[4];
#pragma unroll
  for (int dt = 0; dt < 4; ++dt) o[dt] = {};
  float mr[8], lr[8];
#pragma unroll
  for (int j = 0; j < 8; ++j) { mr[j] = NEG_INF; lr[j] = 0.f; }

  for (int kb = 0; kb <= qb; ++kb) {
    // K tile via async-to-LDS; V tile via registers (transpose) into Vt
    {
      const _Float16* ksrc = Kk + headoff + ((size_t)kb * 64 + lqrow) * kHd + lqco;
      _Float16* kdst = Ks + lqrow * 64 + lqco;
      async_b128(ksrc, kdst);
      async_b128(ksrc + 8, kdst + 8);
      async_b128(ksrc + 16, kdst + 16);
      async_b128(ksrc + 24, kdst + 24);

      const uint4* vs =
          (const uint4*)(V + headoff + ((size_t)kb * 64 + lqrow) * kHd + lqco);
      union { uint4 u[4]; _Float16 h[32]; } vb;
      vb.u[0] = vs[0]; vb.u[1] = vs[1]; vb.u[2] = vs[2]; vb.u[3] = vs[3];
#pragma unroll
      for (int i = 0; i < 32; ++i) Vt[(lqco + i) * 64 + lqrow] = vb.h[i];
    }
    S_WAIT_ASYNC(0x0);   // Q (first iter) + K tile resident in LDS
    __syncthreads();

    // S = Q K^T  (wave rows w*16..w*16+15, keys 0..63)
    v16h aq0 = frag_a_lds(Qs + w * 16 * 64, 64, lane);
    v16h aq1 = frag_a_lds(Qs + w * 16 * 64 + 32, 64, lane);

    float sv[4][8];
#pragma unroll
    for (int ni = 0; ni < 4; ++ni) {
      v8f s = {};
      v16h bk0 = frag_b_lds(Ks + ni * 16 * 64, 64, lane);
      v16h bk1 = frag_b_lds(Ks + ni * 16 * 64 + 32, 64, lane);
      s = wmma16(aq0, bk0, s);
      s = wmma16(aq1, bk1, s);
#pragma unroll
      for (int j = 0; j < 8; ++j) sv[ni][j] = s[j] * 0.125f;  // Hd^-0.5
    }

    if (kb == qb) {  // causal mask on diagonal block
#pragma unroll
      for (int ni = 0; ni < 4; ++ni)
#pragma unroll
        for (int j = 0; j < 8; ++j) {
          const int rl = w * 16 + j + 8 * g;
          const int cl = ni * 16 + c;
          if (cl > rl) sv[ni][j] = NEG_INF;
        }
    }

    // online softmax: reductions across the 16-lane C-layout group
    float tm[8];
#pragma unroll
    for (int j = 0; j < 8; ++j)
      tm[j] = fmaxf(fmaxf(sv[0][j], sv[1][j]), fmaxf(sv[2][j], sv[3][j]));
#pragma unroll
    for (int mk = 1; mk < 16; mk <<= 1)
#pragma unroll
      for (int j = 0; j < 8; ++j)
        tm[j] = fmaxf(tm[j], __shfl_xor(tm[j], mk, 32));

    float alpha[8];
#pragma unroll
    for (int j = 0; j < 8; ++j) {
      const float mn = fmaxf(mr[j], tm[j]);
      alpha[j] = __expf(mr[j] - mn);
      mr[j] = mn;
    }

    float ps[8];
#pragma unroll
    for (int j = 0; j < 8; ++j) ps[j] = 0.f;
#pragma unroll
    for (int ni = 0; ni < 4; ++ni)
#pragma unroll
      for (int j = 0; j < 8; ++j) {
        const float p = __expf(sv[ni][j] - mr[j]);
        sv[ni][j] = p;
        ps[j] += p;
      }
#pragma unroll
    for (int mk = 1; mk < 16; mk <<= 1)
#pragma unroll
      for (int j = 0; j < 8; ++j) ps[j] += __shfl_xor(ps[j], mk, 32);
#pragma unroll
    for (int j = 0; j < 8; ++j) lr[j] = lr[j] * alpha[j] + ps[j];
#pragma unroll
    for (int dt = 0; dt < 4; ++dt)
#pragma unroll
      for (int j = 0; j < 8; ++j) o[dt][j] = o[dt][j] * alpha[j];

    // P (C-layout) -> LDS -> A-layout fragments for P@V
#pragma unroll
    for (int ni = 0; ni < 4; ++ni)
#pragma unroll
      for (int j = 0; j < 8; ++j)
        Ps[(w * 16 + j + 8 * g) * 64 + ni * 16 + c] = (_Float16)sv[ni][j];

    v16h ap0 = frag_a_lds(Ps + w * 16 * 64, 64, lane);
    v16h ap1 = frag_a_lds(Ps + w * 16 * 64 + 32, 64, lane);
#pragma unroll
    for (int dt = 0; dt < 4; ++dt) {
      v16h bv0 = frag_b_lds(Vt + dt * 16 * 64, 64, lane);
      v16h bv1 = frag_b_lds(Vt + dt * 16 * 64 + 32, 64, lane);
      o[dt] = wmma16(ap0, bv0, o[dt]);
      o[dt] = wmma16(ap1, bv1, o[dt]);
    }
    __syncthreads();
  }

  // normalize and emit f16 [B, L, Dim]
  float inv[8];
#pragma unroll
  for (int j = 0; j < 8; ++j) inv[j] = 1.0f / lr[j];
#pragma unroll
  for (int dt = 0; dt < 4; ++dt)
#pragma unroll
    for (int j = 0; j < 8; ++j) {
      const int row = qb * 64 + w * 16 + j + 8 * g;
      const int col = h * kHd + dt * 16 + c;
      Oo[((size_t)b * kL + row) * kDim + col] = (_Float16)(o[dt][j] * inv[j]);
    }
}

// ---------------------------------------------------------------------------
// launch
// ---------------------------------------------------------------------------
extern "C" void kernel_launch(void* const* d_in, const int* in_sizes, int n_in,
                              void* d_out, int out_size, void* d_ws, size_t ws_size,
                              hipStream_t stream) {
  const float* x     = (const float*)d_in[0];  // [4,2048,1024]
  const float* Wqkv  = (const float*)d_in[1];  // [3072,1024]
  const float* Wout  = (const float*)d_in[2];  // [1024,1024]
  float* out         = (float*)d_out;          // [4,2048,1024]

  char* ws = (char*)d_ws;
  size_t off = 0;
  auto carve = [&](size_t bytes) -> void* {
    void* p = ws + off;
    off += (bytes + 255) & ~(size_t)255;
    return p;
  };
  _Float16* xh    = (_Float16*)carve((size_t)kM * kDim * 2);
  _Float16* wqkvh = (_Float16*)carve((size_t)3 * kDim * kDim * 2);
  _Float16* wouth = (_Float16*)carve((size_t)kDim * kDim * 2);
  _Float16* qws   = (_Float16*)carve((size_t)kM * kDim * 2);
  _Float16* kws   = (_Float16*)carve((size_t)kM * kDim * 2);
  _Float16* vws   = (_Float16*)carve((size_t)kM * kDim * 2);
  _Float16* attnh = (_Float16*)carve((size_t)kM * kDim * 2);
  (void)ws_size; (void)in_sizes; (void)n_in; (void)out_size;

  int n4;
  n4 = kM * kDim / 4;
  cvt_f32_f16<<<(n4 + 255) / 256, 256, 0, stream>>>(x, xh, n4);
  n4 = 3 * kDim * kDim / 4;
  cvt_f32_f16<<<(n4 + 255) / 256, 256, 0, stream>>>(Wqkv, wqkvh, n4);
  n4 = kDim * kDim / 4;
  cvt_f32_f16<<<(n4 + 255) / 256, 256, 0, stream>>>(Wout, wouth, n4);

  // qkv = x @ Wqkv^T  -> scatter into q/k/v [B,H,L,Hd]
  gemm_wmma<1><<<dim3(3 * kDim / 128, kM / 128), 256, 0, stream>>>(
      xh, wqkvh, nullptr, qws, kws, vws, kM, 3 * kDim, kDim);

  // causal flash attention
  flash_attn<<<dim3(kL / 64, kHeads, kB), 128, 0, stream>>>(qws, kws, vws, attnh);

  // out = attn @ Wout^T  (fp32)
  gemm_wmma<0><<<dim3(kDim / 128, kM / 128), 256, 0, stream>>>(
      attnh, wouth, out, nullptr, nullptr, nullptr, kM, kDim, kDim);
}
